// TurboQuantV3_11871289607001
// MI455X (gfx1250) — matrix operand order repaired and checked by reading, not compiled, not run on hardware
//
#include <hip/hip_runtime.h>
#include <hip/hip_bf16.h>

typedef __attribute__((ext_vector_type(16))) _Float16 v16h;
typedef __attribute__((ext_vector_type(8)))  float    v8f;

#define D               128
#define ROWS_PER_WAVE   16
#define WAVES_PER_BLOCK 8
#define THREADS         (WAVES_PER_BLOCK * 32)

// LDS-resident R in WMMA B-fragment order:
// ldsB[kk][t][lane] : lane = column t*16+(lane&15), 16 halfs = K-rows kk*32+(lane>>4)*8+{0..7,16..23}
__global__ __launch_bounds__(THREADS)
void tq4_rotquant_kernel(const float* __restrict__ x,
                         const float* __restrict__ R,
                         float* __restrict__ outPacked,  // N*64
                         float* __restrict__ outNorm,    // N
                         float* __restrict__ outMin,     // N
                         float* __restrict__ outScale,   // N
                         int nStripes)
{
    __shared__ v16h ldsB[4][8][32];   // 4*8*32*32B = 32 KB

    // ---- stage R -> LDS as f16 fragments (once per block) ----
    for (int chunk = threadIdx.x; chunk < 4 * 8 * 32; chunk += THREADS) {
        int lane_ = chunk & 31;
        int t     = (chunk >> 5) & 7;
        int kk    = chunk >> 8;
        const float* src = R + (size_t)(kk * 32 + (lane_ >> 4) * 8) * D + t * 16 + (lane_ & 15);
        v16h v;
        #pragma unroll
        for (int j = 0; j < 16; ++j) v[j] = (_Float16)src[(size_t)(j + (j >> 3) * 8) * D];
        ldsB[kk][t][lane_] = v;
    }
    __syncthreads();

    const int lane = threadIdx.x & 31;
    const int nl   = lane & 15;        // N-lane / M-row index within half
    const int hi   = lane >> 4;        // which half of the wave
    const int waveId = blockIdx.x * WAVES_PER_BLOCK + (threadIdx.x >> 5);
    const int nWaves = gridDim.x * WAVES_PER_BLOCK;

    for (int stripe = waveId; stripe < nStripes; stripe += nWaves) {
        const int rowBase = stripe * ROWS_PER_WAVE;
        const int myRow   = rowBase + nl;                 // A-fragment row (M = nl)
        const float* xr   = x + (size_t)myRow * D;

        // prefetch next stripe's row (4 x 128B lines per row, covered wave-wide)
        const int nextStripe = stripe + nWaves;
        if (nextStripe < nStripes) {
            const float* xn = x + ((size_t)nextStripe * ROWS_PER_WAVE + nl) * D;
            #pragma unroll
            for (int kk = 0; kk < 4; ++kk)
                __builtin_prefetch(xn + kk * 32, 0, 1);
        }

        // ---- load this lane's half of row M=nl (ISA A-layout, b128 chunks),
        //      accumulate sum of squares, convert RAW values to f16 ----
        float ss = 0.f;
        v16h af[4];
        #pragma unroll
        for (int kk = 0; kk < 4; ++kk) {
            const int k0 = kk * 32 + hi * 8;
            union { float4 v4[4]; float f[16]; } ld;
            ld.v4[0] = *(const float4*)(xr + k0);
            ld.v4[1] = *(const float4*)(xr + k0 + 4);
            ld.v4[2] = *(const float4*)(xr + k0 + 16);
            ld.v4[3] = *(const float4*)(xr + k0 + 20);
            #pragma unroll
            for (int j = 0; j < 16; ++j) {
                const float v = ld.f[j];
                ss = fmaf(v, v, ss);
                af[kk][j] = (_Float16)v;          // raw (unnormalized) — see note
            }
        }
        ss += __shfl_xor(ss, 16, 32);             // combine the two half-rows
        const float norm = sqrtf(ss);
        const float inv  = 1.0f / fmaxf(norm, 1e-8f);
        // NOTE: (x/||x||)@R == (x@R)/||x||, and the 4-bit index is invariant to a
        // positive row scale, so we rotate/quantize raw and scale only min/scale.

        // ---- rotated_raw = x @ R : 8 column tiles x 4 K-steps of WMMA ----
        v8f acc[8];
        #pragma unroll
        for (int t = 0; t < 8; ++t) {
            v8f c = {};
            #pragma unroll
            for (int kk = 0; kk < 4; ++kk) {
                v16h b = ldsB[kk][t][lane];
                c = __builtin_amdgcn_wmma_f32_16x16x32_f16(
                        false, af[kk], false, b, (short)0, c, false, false);
            }
            acc[t] = c;
        }

        // C/D layout: lane holds rows m = hi*8 + v (v = VGPR 0..7), column n = t*16 + nl
        // ---- per-row min/max across 128 columns (raw domain) ----
        #pragma unroll
        for (int v = 0; v < 8; ++v) {
            float mn = acc[0][v], mx = acc[0][v];
            #pragma unroll
            for (int t = 1; t < 8; ++t) { mn = fminf(mn, acc[t][v]); mx = fmaxf(mx, acc[t][v]); }
            #pragma unroll
            for (int mask = 1; mask < 16; mask <<= 1) {   // stays inside the 16-lane half
                mn = fminf(mn, __shfl_xor(mn, mask, 32));
                mx = fmaxf(mx, __shfl_xor(mx, mask, 32));
            }

            const float range = mx - mn;
            const float rq    = 15.0f / fmaxf(range, 1e-30f);
            const int   row   = rowBase + hi * 8 + v;

            // ---- quantize + nibble-pack (byte value kept in float: exact) ----
            #pragma unroll
            for (int t = 0; t < 8; ++t) {
                float q  = fminf(fmaxf(rintf((acc[t][v] - mn) * rq), 0.f), 15.f);
                float qo = __shfl_xor(q, 1, 32);          // neighbor column's nibble
                if ((nl & 1) == 0)
                    outPacked[(size_t)row * 64 + t * 8 + (nl >> 1)] = fmaf(q, 16.f, qo);
            }

            // writer lane nl == hi*8+v owns this row's inv -> no broadcast needed
            if (nl == hi * 8 + v) {
                outMin[row]   = mn * inv;
                outScale[row] = fmaxf(range * inv, 1e-8f) * (1.0f / 15.0f);
            }
        }
        if (hi == 0) outNorm[myRow] = norm;
    }
}

extern "C" void kernel_launch(void* const* d_in, const int* in_sizes, int n_in,
                              void* d_out, int out_size, void* d_ws, size_t ws_size,
                              hipStream_t stream)
{
    const float* x = (const float*)d_in[0];   // (B,H,S,D) f32
    const float* R = (const float*)d_in[1];   // (128,128) f32

    const int N = in_sizes[0] / D;            // number of rows = B*H*S
    float* base      = (float*)d_out;
    float* outPacked = base;                        // N*64
    float* outNorm   = base + (size_t)N * 64;       // N
    float* outMin    = base + (size_t)N * 65;       // N
    float* outScale  = base + (size_t)N * 66;       // N

    const int nStripes = N / ROWS_PER_WAVE;
    int blocks = 1024;                               // grid-stride; amortizes R->LDS staging
    if (blocks * WAVES_PER_BLOCK > nStripes)
        blocks = (nStripes + WAVES_PER_BLOCK - 1) / WAVES_PER_BLOCK;
    if (blocks < 1) blocks = 1;

    tq4_rotquant_kernel<<<blocks, THREADS, 0, stream>>>(
        x, R, outPacked, outNorm, outMin, outScale, nStripes);
}
